// YOLOv8DetectionLoss_23124103922075
// MI455X (gfx1250) — compile-verified, matrix operand order checked by reading, not run on hardware
//
#include <hip/hip_runtime.h>
#include <hip/hip_bf16.h>
#include <math.h>

#define B_N 16
#define M_N 64
#define A_N 8400
#define C_N 80
#define TOPK_N 13
#define CHUNK 1024

typedef __attribute__((ext_vector_type(16))) _Float16 v16h;
typedef __attribute__((ext_vector_type(8)))  float    v8f;
typedef unsigned int u32x4 __attribute__((ext_vector_type(4)));
typedef int          i32x8 __attribute__((ext_vector_type(8)));
typedef int          i32x4 __attribute__((ext_vector_type(4)));

// ---- TDM availability / arity detection (compile-safe fallbacks) ----
#if defined(__has_include)
#  if __has_include(<hip/amd_detail/amd_gfx1250_TDM.h>)
#    define CDNA5_TDM_6ARG 1
#  endif
#endif
#ifndef CDNA5_TDM_6ARG
#  if defined(__clang_major__) && (__clang_major__ >= 23)
#    define CDNA5_TDM_6ARG 1
#  else
#    define CDNA5_TDM_6ARG 0
#  endif
#endif

#if defined(__has_builtin)
#  if __has_builtin(__builtin_amdgcn_tensor_load_to_lds) && __has_builtin(__builtin_amdgcn_s_wait_tensorcnt)
#    define HAVE_TDM 1
#  endif
#endif

#ifdef HAVE_TDM
// 1-D contiguous copy of n dwords (global -> LDS) via the Tensor Data Mover.
__device__ __forceinline__ void tdm_load_1d(unsigned lds_off, unsigned long long ga, unsigned n)
{
  u32x4 g0;
  g0[0] = 1u;                                  // count=1 (valid user descriptor)
  g0[1] = lds_off;                             // LDS byte address
  g0[2] = (unsigned)ga;                        // global_addr[31:0]
  g0[3] = ((unsigned)(ga >> 32) & 0x01FFFFFFu) | 0x80000000u; // addr[56:32] | type=2
  i32x8 g1;
  g1[0] = 0x00020000;                          // workgroup_mask=0, data_size=2 (4 bytes)
  g1[1] = (int)((n & 0xFFFFu) << 16);          // tensor_dim0[15:0] @ bits 63:48
  g1[2] = (int)(((n >> 16) & 0xFFFFu) | (1u << 16)); // tensor_dim0 hi | tensor_dim1=1
  g1[3] = (int)((n & 0xFFFFu) << 16);          // tile_dim0 = n @ bits 127:112
  g1[4] = 0;                                   // tile_dim1=0 (unused), tile_dim2=0
  g1[5] = (int)n;                              // tensor_dim0_stride lo
  g1[6] = 0;
  g1[7] = 0;
  i32x4 g2; g2[0] = 1; g2[1] = 1; g2[2] = 0; g2[3] = 0;
  i32x4 g3; g3[0] = 0; g3[1] = (1 << 16); g3[2] = 0; g3[3] = 0;
#if CDNA5_TDM_6ARG
  i32x8 g4; g4[0]=0;g4[1]=0;g4[2]=0;g4[3]=0;g4[4]=0;g4[5]=0;g4[6]=0;g4[7]=0;
  __builtin_amdgcn_tensor_load_to_lds(g0, g1, g2, g3, g4, 0);
#else
  __builtin_amdgcn_tensor_load_to_lds(g0, g1, g2, g3, 0);
#endif
}
#endif

// CIoU (b1 = gt, b2 = pred), clipped at 0.  at1/at2 = atan(w/h) precomputed.
__device__ __forceinline__ float ciou_clip(float gx1, float gy1, float gx2, float gy2,
                                           float w1, float h1, float at1,
                                           float px1, float py1, float px2, float py2,
                                           float at2)
{
  float w2 = px2 - px1, h2 = py2 - py1 + 1e-7f;
  float iw = fmaxf(fminf(gx2, px2) - fmaxf(gx1, px1), 0.f);
  float ih = fmaxf(fminf(gy2, py2) - fmaxf(gy1, py1), 0.f);
  float inter = iw * ih;
  float uni = w1 * h1 + w2 * h2 - inter + 1e-7f;
  float iou = inter / uni;
  float cw = fmaxf(gx2, px2) - fminf(gx1, px1);
  float ch = fmaxf(gy2, py2) - fminf(gy1, py1);
  float c2 = cw * cw + ch * ch + 1e-7f;
  float dx = px1 + px2 - gx1 - gx2, dy = py1 + py2 - gy1 - gy2;
  float rho2 = 0.25f * (dx * dx + dy * dy);
  float dat = at2 - at1;
  float v = 0.405284734569351086f * dat * dat;   // 4/pi^2
  float alpha = v / (v - iou + 1.0000001f);
  return fmaxf(iou - (rho2 / c2 + v * alpha), 0.f);
}

__global__ void k_zero(unsigned int* __restrict__ p, int n)
{
  int i = blockIdx.x * 256 + threadIdx.x;
  if (i < n) p[i] = 0u;
}

// ---------------------------------------------------------------------------
// K1: one block per (b, m).  Builds the align row over all 8400 anchors
// (scores via one-hot WMMA, boxes TDM-staged in LDS), then exact top-13 with
// jax.top_k tie-breaking, then scatters bit m into topk_bits for valid GTs.
// ---------------------------------------------------------------------------
__global__ __launch_bounds__(256)
void k_topk(const float* __restrict__ pd_scores, const float* __restrict__ pd_boxes,
            const float* __restrict__ anchors,   const int* __restrict__ gt_classes,
            const float* __restrict__ gt_boxes,  const int* __restrict__ gt_mask,
            unsigned long long* __restrict__ topk_bits)
{
  __shared__ float s_align[A_N];        // 33.6 KB
  __shared__ float s_box[CHUNK * 4];    // 16 KB staged pd_boxes chunk
  __shared__ float s_rv[256];
  __shared__ int   s_ri[256];
  __shared__ int   s_win[TOPK_N];

  const int b    = blockIdx.x >> 6;
  const int m    = blockIdx.x & 63;
  const int tid  = threadIdx.x;
  const int lane = tid & 31;
  const int wave = tid >> 5;

  const float gx1 = gt_boxes[(b * M_N + m) * 4 + 0];
  const float gy1 = gt_boxes[(b * M_N + m) * 4 + 1];
  const float gx2 = gt_boxes[(b * M_N + m) * 4 + 2];
  const float gy2 = gt_boxes[(b * M_N + m) * 4 + 3];
  const int   cls   = gt_classes[b * M_N + m];
  const int   valid = gt_mask[b * M_N + m];
  const float w1 = gx2 - gx1, h1 = gy2 - gy1 + 1e-7f;
  const float at1 = atanf(w1 / h1);

  // One-hot A matrices (row M=0 only; lives in lanes 0 and 16) are invariant
  // across all tiles: build the three K-slabs once, branchlessly.
  // Lane 0 covers K = {0..7, 16..23}; lane 16 covers K = {8..15, 24..31}
  // (+ t*32 class offset).  Other lanes: impossible sentinel -> all zeros.
  v16h va0, va1, va2;
  {
    const int sel = (lane == 0) ? 0 : ((lane == 16) ? 8 : 4096);
    #pragma unroll
    for (int h = 0; h < 16; ++h) {
      const int K = (h & 7) + ((h >> 3) << 4) + sel;
      va0[h] = (K == cls)      ? (_Float16)1.0f : (_Float16)0.0f;
      va1[h] = (K == cls - 32) ? (_Float16)1.0f : (_Float16)0.0f;
      va2[h] = (K == cls - 64) ? (_Float16)1.0f : (_Float16)0.0f;
    }
  }

  const float* pb_b = pd_boxes  + (size_t)b * A_N * 4;
  const float* ps_b = pd_scores + (size_t)b * A_N * C_N;

  const int koff = (lane >= 16) ? 8 : 0;
  const int n    = lane & 15;

  for (int chunk = 0; chunk < (A_N + CHUNK - 1) / CHUNK; ++chunk) {
    const int cbase = chunk * CHUNK;
    const int cnt   = min(CHUNK, A_N - cbase);
    __syncthreads();                       // previous chunk consumers done
#ifdef HAVE_TDM
    if (wave == 0) {
      tdm_load_1d((unsigned)(size_t)s_box,
                  (unsigned long long)(size_t)(pb_b + (size_t)cbase * 4),
                  (unsigned)(cnt * 4));
      __builtin_amdgcn_s_wait_tensorcnt(0);
    }
#else
    for (int j = tid; j < cnt * 4; j += 256)
      s_box[j] = pb_b[(size_t)cbase * 4 + j];
#endif
    __syncthreads();

    const int ntiles = (cnt + 15) >> 4;    // 16-anchor tiles in this chunk
    for (int it = 0; it < (ntiles + 7) / 8; ++it) {
      const int tl = it * 8 + wave;        // wave-uniform -> EXEC full in active waves
      if (tl >= ntiles) continue;
      const int abase = cbase + tl * 16;

      // ---- scores S(row0, 16 anchors) = onehot(1x80) x P(80x16) via WMMA f16 ----
      const float* prow = ps_b + (size_t)(abase + n) * C_N + koff;
      v8f acc = {};
      #pragma unroll
      for (int t = 0; t < 3; ++t) {        // K = 96 (80 classes zero-padded)
        // Runs are in-bounds purely as a function of t (compile-time after
        // unroll): t<2 -> both 8-class runs valid; t==2 -> second run is the
        // 80..95 zero pad.  No lane-divergent guards.
        const float4 r0 = *(const float4*)(prow + t * 32);
        const float4 r1 = *(const float4*)(prow + t * 32 + 4);
        float4 r2, r3;
        if (t < 2) {
          r2 = *(const float4*)(prow + t * 32 + 16);
          r3 = *(const float4*)(prow + t * 32 + 20);
        } else {
          r2 = make_float4(0.f, 0.f, 0.f, 0.f);
          r3 = make_float4(0.f, 0.f, 0.f, 0.f);
        }
        v16h vb;
        vb[0]=(_Float16)r0.x;  vb[1]=(_Float16)r0.y;  vb[2]=(_Float16)r0.z;  vb[3]=(_Float16)r0.w;
        vb[4]=(_Float16)r1.x;  vb[5]=(_Float16)r1.y;  vb[6]=(_Float16)r1.z;  vb[7]=(_Float16)r1.w;
        vb[8]=(_Float16)r2.x;  vb[9]=(_Float16)r2.y;  vb[10]=(_Float16)r2.z; vb[11]=(_Float16)r2.w;
        vb[12]=(_Float16)r3.x; vb[13]=(_Float16)r3.y; vb[14]=(_Float16)r3.z; vb[15]=(_Float16)r3.w;
        const v16h va = (t == 0) ? va0 : ((t == 1) ? va1 : va2);
        acc = __builtin_amdgcn_wmma_f32_16x16x32_f16(false, va, false, vb,
                                                     (short)0, acc, false, false);
      }
      const float sc = acc[0];             // lanes 0..15: score of anchor abase+lane

      if (lane < 16) {
        const int a = abase + lane;
        const float ax = anchors[a * 2 + 0], ay = anchors[a * 2 + 1];
        const float* pbx = &s_box[(a - cbase) * 4];
        const float px1 = pbx[0], py1 = pbx[1], px2 = pbx[2], py2 = pbx[3];
        const bool msk = fminf(fminf(ax - gx1, ay - gy1), fminf(gx2 - ax, gy2 - ay)) > 1e-9f;
        float iou = 0.f, s = 0.f;
        if (msk) {
          const float w2 = px2 - px1, h2 = py2 - py1 + 1e-7f;
          iou = ciou_clip(gx1, gy1, gx2, gy2, w1, h1, at1, px1, py1, px2, py2, atanf(w2 / h2));
          s = sc;
        }
        const float i2 = iou * iou;
        s_align[a] = s * i2 * i2 * i2;     // score^1 * iou^6
      }
    }
  }
  __syncthreads();

  // top-13 via 13 argmax rounds; ties -> lowest index (matches jax.lax.top_k)
  for (int r = 0; r < TOPK_N; ++r) {
    float bv = -2.f; int bi = 0;
    for (int a = tid; a < A_N; a += 256) { // ascending stride keeps first max
      const float v = s_align[a];
      if (v > bv) { bv = v; bi = a; }
    }
    s_rv[tid] = bv; s_ri[tid] = bi;
    __syncthreads();
    for (int s = 128; s > 0; s >>= 1) {
      if (tid < s) {
        const float ov = s_rv[tid + s]; const int oi = s_ri[tid + s];
        if (ov > s_rv[tid] || (ov == s_rv[tid] && oi < s_ri[tid])) { s_rv[tid] = ov; s_ri[tid] = oi; }
      }
      __syncthreads();
    }
    if (tid == 0) { const int w = s_ri[0]; s_win[r] = w; s_align[w] = -1.f; }
    __syncthreads();
  }

  if (valid && tid < TOPK_N)
    atomicOr(&topk_bits[(size_t)b * A_N + s_win[tid]], 1ull << m);
}

// ---------------------------------------------------------------------------
// K2: one thread per (b, a).  Resolves positives + duplicates, emits boxes /
// fg, per-anchor align value, and per-(b,m) row maxima via int atomicMax.
// ---------------------------------------------------------------------------
__global__ __launch_bounds__(256)
void k_assign(const float* __restrict__ pd_scores, const float* __restrict__ pd_boxes,
              const float* __restrict__ anchors,   const int* __restrict__ gt_classes,
              const float* __restrict__ gt_boxes,  const int* __restrict__ gt_mask,
              const unsigned long long* __restrict__ topk_bits,
              int* __restrict__ info, float* __restrict__ alignval,
              int* __restrict__ rowMaxAlign, int* __restrict__ rowMaxIou,
              float* __restrict__ out_boxes, float* __restrict__ out_fg)
{
  __shared__ float s_gt[M_N * 4];
  __shared__ float s_at[M_N];
  __shared__ int   s_cls[M_N];
  const int b   = blockIdx.y;
  const int tid = threadIdx.x;
  if (tid < M_N * 4) s_gt[tid] = gt_boxes[b * M_N * 4 + tid];
  if (tid < M_N) {
    s_cls[tid] = gt_classes[b * M_N + tid];
    const float ww = gt_boxes[(b * M_N + tid) * 4 + 2] - gt_boxes[(b * M_N + tid) * 4 + 0];
    const float hh = gt_boxes[(b * M_N + tid) * 4 + 3] - gt_boxes[(b * M_N + tid) * 4 + 1] + 1e-7f;
    s_at[tid] = atanf(ww / hh);
  }
  __syncthreads();

  const int a = blockIdx.x * 256 + tid;
  if (a >= A_N) return;
  const size_t ba = (size_t)b * A_N + a;

  const unsigned long long bits = topk_bits[ba];
  const float ax = anchors[a * 2 + 0], ay = anchors[a * 2 + 1];
  const float4 pbv = *(const float4*)(pd_boxes + ba * 4);
  const float px1 = pbv.x, py1 = pbv.y, px2 = pbv.z, py2 = pbv.w;
  const float w2 = px2 - px1, h2 = py2 - py1 + 1e-7f;
  const float at2 = atanf(w2 / h2);

  unsigned long long posbits = 0ull;
  float best_iou = -1.f; int best_m = 0;
  for (int m = 0; m < M_N; ++m) {
    const float gx1 = s_gt[m * 4 + 0], gy1 = s_gt[m * 4 + 1];
    const float gx2 = s_gt[m * 4 + 2], gy2 = s_gt[m * 4 + 3];
    const bool msk = fminf(fminf(ax - gx1, ay - gy1), fminf(gx2 - ax, gy2 - ay)) > 1e-9f;
    float iou = 0.f;
    if (msk) {
      const float w1 = gx2 - gx1, h1 = gy2 - gy1 + 1e-7f;
      iou = ciou_clip(gx1, gy1, gx2, gy2, w1, h1, s_at[m], px1, py1, px2, py2, at2);
    }
    if (iou > best_iou) { best_iou = iou; best_m = m; }   // first-max, like jnp.argmax
    if (((bits >> m) & 1ull) && msk) posbits |= (1ull << m);
  }
  int fg = __popcll(posbits);
  if (fg > 1) { posbits = 1ull << best_m; fg = 1; }        // duplicate resolution
  const int m_star = posbits ? (__ffsll(posbits) - 1) : 0; // argmax of boolean

  *(float4*)(out_boxes + ba * 4) = make_float4(s_gt[m_star * 4 + 0], s_gt[m_star * 4 + 1],
                                               s_gt[m_star * 4 + 2], s_gt[m_star * 4 + 3]);
  out_fg[ba] = fg ? 1.f : 0.f;

  float av = 0.f;
  if (fg) {
    const float gx1 = s_gt[m_star * 4 + 0], gy1 = s_gt[m_star * 4 + 1];
    const float gx2 = s_gt[m_star * 4 + 2], gy2 = s_gt[m_star * 4 + 3];
    const bool msk = fminf(fminf(ax - gx1, ay - gy1), fminf(gx2 - ax, gy2 - ay)) > 1e-9f;
    float iv = 0.f, sv = 0.f;
    if (msk) {
      const float w1 = gx2 - gx1, h1 = gy2 - gy1 + 1e-7f;
      iv = ciou_clip(gx1, gy1, gx2, gy2, w1, h1, s_at[m_star], px1, py1, px2, py2, at2);
      sv = pd_scores[ba * C_N + s_cls[m_star]];           // exact f32 gather for norm path
    }
    const float i2 = iv * iv;
    av = sv * i2 * i2 * i2;
    atomicMax(&rowMaxAlign[b * M_N + m_star], __float_as_int(av)); // values >= 0
    atomicMax(&rowMaxIou[b * M_N + m_star], __float_as_int(iv));
  }
  info[ba] = m_star | (fg << 8);
  alignval[ba] = av;
}

// ---------------------------------------------------------------------------
// K3: one thread per (b, a): writes the 80-class one-hot row scaled by norm.
// ---------------------------------------------------------------------------
__global__ __launch_bounds__(256)
void k_scores(const int* __restrict__ gt_classes, const int* __restrict__ info,
              const float* __restrict__ alignval, const int* __restrict__ rowMaxAlign,
              const int* __restrict__ rowMaxIou, float* __restrict__ out_scores)
{
  const int idx = blockIdx.x * 256 + threadIdx.x;
  if (idx >= B_N * A_N) return;
  const int b  = idx / A_N;
  const int inf = info[idx];
  const int m  = inf & 0xFF;
  const int fg = inf >> 8;
  float norm = 0.f; int tc = 0;
  if (fg) {
    const float ma = __int_as_float(rowMaxAlign[b * M_N + m]);
    const float mi = __int_as_float(rowMaxIou[b * M_N + m]);
    norm = alignval[idx] * mi / (ma + 1e-9f);
    tc = max(gt_classes[b * M_N + m], 0);
  }
  float* row = out_scores + (size_t)idx * C_N;
  const float4 z = {0.f, 0.f, 0.f, 0.f};
  #pragma unroll
  for (int j = 0; j < C_N / 4; ++j) ((float4*)row)[j] = z;
  if (fg) row[tc] = norm;
}

extern "C" void kernel_launch(void* const* d_in, const int* in_sizes, int n_in,
                              void* d_out, int out_size, void* d_ws, size_t ws_size,
                              hipStream_t stream)
{
  (void)in_sizes; (void)n_in; (void)out_size; (void)ws_size;
  const float* pd_scores = (const float*)d_in[0];
  const float* pd_boxes  = (const float*)d_in[1];
  const float* anchors   = (const float*)d_in[2];
  const int*   gt_cls    = (const int*)d_in[3];
  const float* gt_boxes  = (const float*)d_in[4];
  const int*   gt_mask   = (const int*)d_in[5];

  float* out        = (float*)d_out;
  float* out_scores = out;                                   // B*A*80
  float* out_boxes  = out + (size_t)B_N * A_N * C_N;         // B*A*4
  float* out_fg     = out_boxes + (size_t)B_N * A_N * 4;     // B*A

  // ws layout: [topk_bits u64 B*A][rowMaxAlign i32 B*M][rowMaxIou i32 B*M]
  //            [info i32 B*A][alignval f32 B*A]   (~2.16 MB)
  unsigned long long* topk_bits = (unsigned long long*)d_ws;
  int*   rowMaxAlign = (int*)(topk_bits + (size_t)B_N * A_N);
  int*   rowMaxIou   = rowMaxAlign + B_N * M_N;
  int*   info        = rowMaxIou + B_N * M_N;
  float* alignval    = (float*)(info + (size_t)B_N * A_N);

  const int nzero = B_N * A_N * 2 + B_N * M_N * 2;  // topk_bits + both row-max arrays
  k_zero<<<(nzero + 255) / 256, 256, 0, stream>>>((unsigned int*)d_ws, nzero);

  k_topk<<<B_N * M_N, 256, 0, stream>>>(pd_scores, pd_boxes, anchors, gt_cls,
                                        gt_boxes, gt_mask, topk_bits);

  dim3 ga((A_N + 255) / 256, B_N);
  k_assign<<<ga, 256, 0, stream>>>(pd_scores, pd_boxes, anchors, gt_cls, gt_boxes, gt_mask,
                                   topk_bits, info, alignval, rowMaxAlign, rowMaxIou,
                                   out_boxes, out_fg);

  k_scores<<<(B_N * A_N + 255) / 256, 256, 0, stream>>>(gt_cls, info, alignval,
                                                        rowMaxAlign, rowMaxIou, out_scores);
}
